// SoftmaxAttention_55611236549123
// MI455X (gfx1250) — matrix operand order repaired and checked by reading, not compiled, not run on hardware
//
#include <hip/hip_runtime.h>
#include <hip/hip_bf16.h>

#define B_ 2
#define H_ 16
#define S_ 2048
#define D_ 64

typedef __bf16 v16bf __attribute__((ext_vector_type(16)));
typedef float  v8f   __attribute__((ext_vector_type(8)));
typedef unsigned int u32x4 __attribute__((ext_vector_type(4)));
typedef int          i32x4 __attribute__((ext_vector_type(4)));
typedef int          i32x8 __attribute__((ext_vector_type(8)));

union BF16x16 {
    v16bf    v;
    __bf16   e[16];
    uint32_t u[8];
};

// ---------------- TDM descriptor helpers (CDNA5 ISA ch. 8 D# layout) ----------
// Group0: [1:0]=count=1, [63:32]=lds_addr, [120:64]=global_addr, [127:126]=type(2)
__device__ __forceinline__ u32x4 tdm_g0(uint32_t lds_addr, unsigned long long gaddr) {
    u32x4 g;
    g.x = 1u;
    g.y = lds_addr;
    g.z = (uint32_t)gaddr;
    g.w = (uint32_t)(gaddr >> 32) | (2u << 30);
    return g;
}

// Group1: data_size=2B, pad_enable, pad_amount=1 DWORD, 2-D tile
// [17:16]=data_size(1), [20]=pad_en, [24:22]=pad_interval, [79:48]=tensor_dim0,
// [111:80]=tensor_dim1, [127:112]=tile_dim0, [143:128]=tile_dim1,
// [207:160]=tensor_dim0_stride
__device__ __forceinline__ i32x8 tdm_g1(uint32_t tensor_d0, uint32_t tensor_d1,
                                        uint32_t tile_d0, uint32_t tile_d1,
                                        unsigned long long stride0, uint32_t pad_code) {
    union { unsigned long long q[4]; i32x8 v; } u;
    u.q[0] = (1ull << 16) | (1ull << 20) | ((unsigned long long)pad_code << 22)
           | ((unsigned long long)(tensor_d0 & 0xFFFFu) << 48);
    u.q[1] = ((unsigned long long)(tensor_d0 >> 16) & 0xFFFFull)
           | ((unsigned long long)tensor_d1 << 16)
           | ((unsigned long long)tile_d0 << 48);
    u.q[2] = (unsigned long long)tile_d1 | (stride0 << 32);
    u.q[3] = (stride0 >> 32) & 0xFFFFull;
    return u.v;
}

__device__ __forceinline__ void tdm_load(u32x4 g0, i32x8 g1) {
#if __clang_major__ >= 23
    i32x4 z4 = {}; i32x8 z8 = {};
    __builtin_amdgcn_tensor_load_to_lds(g0, g1, z4, z4, z8, 0);
#else
    i32x4 z4 = {};
    __builtin_amdgcn_tensor_load_to_lds(g0, g1, z4, z4, 0);
#endif
}

// ---------------- prepass: elementwise fp32 -> bf16 for K and V ---------------
__global__ __launch_bounds__(256) void SoftmaxAttention_prep_kernel(
    const float* __restrict__ K, const float* __restrict__ V,
    __bf16* __restrict__ Kb, __bf16* __restrict__ Vb)
{
    const int tid = threadIdx.x;
    #pragma unroll
    for (int it = 0; it < 4; ++it) {
        const size_t i = (size_t)blockIdx.x * 1024 + tid + it * 256;  // float4 idx
        const size_t g = i * 4;
        float4 kx = *(const float4*)(K + g);
        union { __bf16 e[4]; unsigned long long u; } pk;
        pk.e[0] = (__bf16)kx.x; pk.e[1] = (__bf16)kx.y;
        pk.e[2] = (__bf16)kx.z; pk.e[3] = (__bf16)kx.w;
        *(unsigned long long*)(Kb + g) = pk.u;
        float4 vx = *(const float4*)(V + g);
        union { __bf16 e[4]; unsigned long long u; } pv;
        pv.e[0] = (__bf16)vx.x; pv.e[1] = (__bf16)vx.y;
        pv.e[2] = (__bf16)vx.z; pv.e[3] = (__bf16)vx.w;
        *(unsigned long long*)(Vb + g) = pv.u;
    }
}

// ---------------- main flash-attention kernel --------------------------------
// Computes S^T = K.Q^T so that queries live in lanes (per-lane softmax) and
// P lands directly in the A-fragment layout of the P.V WMMA (no transpose).
__global__ __launch_bounds__(128) void SoftmaxAttention_55611236549123_kernel(
    const float* __restrict__ Q, const __bf16* __restrict__ Kbg,
    const __bf16* __restrict__ Vbg, const float* __restrict__ mask,
    float* __restrict__ Out)
{
    // Double-buffered TDM destinations; TDM pad (1 DWORD per 32 DWORDs)
    // produces the 66-bf16 padded row stride.
    __shared__ __bf16 kT[2][32][66];        // [buf][key][d]
    __shared__ __bf16 vT[2][32][66];        // [buf][key][d]
    __shared__ float  penL[S_];             // additive mask penalty, exp2 domain

    const int tid  = threadIdx.x;
    const int lane = tid & 31;
    const int wave = tid >> 5;
    const int half = lane >> 4;
    const int l15  = lane & 15;

    const int qBlocks = S_ / 64;
    const int wg = blockIdx.x;
    const int bh = wg / qBlocks;
    const int qb = wg % qBlocks;
    const int b  = bh / H_;
    const int qBase = qb * 64 + wave * 16;

    // exp2-domain constants (v_exp_f32 is base-2 natively)
    const float LOG2E  = 1.44269504088896341f;
    const float SCALE2 = 0.125f * LOG2E;          // 1/sqrt(64) * log2(e)
    const float PEN2   = 1000000.0f * LOG2E;

    const float* maskRow = mask + (size_t)b * S_;

    // ---- penalty table into LDS (visible before first compute barrier) ----
    for (int i = tid; i < S_; i += 128)
        penL[i] = -PEN2 * (1.0f - maskRow[i]);

    // ---- Q^T as two bf16 B-fragments: B[k=d_local][n=query] ----
    // lane holds query l15; elem j = d_local 8*half+j (j<8), 16+8*half+(j-8) (j>=8)
    v16bf qbf[2];
    {
        const float* Qb = Q + ((size_t)bh * S_ + qBase) * D_;
        const int dcol = 8 * half;
        #pragma unroll
        for (int dh = 0; dh < 2; ++dh) {
            BF16x16 u;
            #pragma unroll
            for (int j = 0; j < 16; ++j)
                u.e[j] = (__bf16)Qb[(size_t)l15 * D_ + 32 * dh + dcol + (j < 8 ? j : 8 + j)];
            qbf[dh] = u.v;
        }
    }

    // Constant all-ones B fragment: row-sum of P via the matrix pipe.
    BF16x16 ones;
    #pragma unroll
    for (int j = 0; j < 8; ++j) ones.u[j] = 0x3F803F80u;   // two bf16 1.0

    v8f acc[4];
    #pragma unroll
    for (int t = 0; t < 4; ++t) { v8f z = {}; acc[t] = z; }
    v8f lacc; { v8f z = {}; lacc = z; }     // running row-sum accumulator
    float mrun = -3.0e38f;                  // per-lane (= per-query) running max

    // ---- TDM setup (uniform): both tensors are [bh][key][d] bf16 ----
    const unsigned long long kA =
        (unsigned long long)(uintptr_t)(Kbg + (size_t)bh * S_ * D_);
    const unsigned long long vA =
        (unsigned long long)(uintptr_t)(Vbg + (size_t)bh * S_ * D_);
    const uint32_t kL[2] = { (uint32_t)(uintptr_t)&kT[0][0][0],
                             (uint32_t)(uintptr_t)&kT[1][0][0] };
    const uint32_t vL[2] = { (uint32_t)(uintptr_t)&vT[0][0][0],
                             (uint32_t)(uintptr_t)&vT[1][0][0] };
    // tile: 32 rows(key) x 64 d, row=128B -> pad_interval code 4 (32 DW)
    const i32x8 g1KV = tdm_g1(D_, S_, D_, 32, (unsigned long long)D_, 4);

    if (wave == 0) {  // prologue: tile 0 into buffer 0
        tdm_load(tdm_g0(kL[0], kA), g1KV);
        tdm_load(tdm_g0(vL[0], vA), g1KV);
    }

    const int NIT = S_ / 32;
    for (int it = 0; it < NIT; ++it) {
        const int buf = it & 1;
        const int kb  = it * 32;

        __syncthreads();  // everyone done reading buf^1 (tile it-1)
        if (wave == 0) {
            if (it + 1 < NIT) {
                const int nb = buf ^ 1;
                const unsigned long long off = 2ull * (unsigned long long)((it + 1) * 32 * D_);
                tdm_load(tdm_g0(kL[nb], kA + off), g1KV);
                tdm_load(tdm_g0(vL[nb], vA + off), g1KV);
                __builtin_amdgcn_s_wait_tensorcnt((short)2);  // tile `it` complete
            } else {
                __builtin_amdgcn_s_wait_tensorcnt((short)0);
            }
        }
        __syncthreads();  // tile `buf` visible to all waves

        // ---- S^T tiles: t=0 -> keys 0-15, t=1 -> keys 16-31 (A=K, B=Q^T) ----
        v8f s[2];
        #pragma unroll
        for (int t = 0; t < 2; ++t) {
            v8f c = {};
            #pragma unroll
            for (int dh = 0; dh < 2; ++dh) {
                // A-frag of K: lane = key row t*16+l15; elems = d runs per half
                BF16x16 ka;
                const uint32_t* pa =
                    (const uint32_t*)&kT[buf][t * 16 + l15][32 * dh + 8 * half];
                const uint32_t* pb =
                    (const uint32_t*)&kT[buf][t * 16 + l15][32 * dh + 16 + 8 * half];
                #pragma unroll
                for (int j = 0; j < 4; ++j) { ka.u[j] = pa[j]; ka.u[4 + j] = pb[j]; }
                c = __builtin_amdgcn_wmma_f32_16x16x32_bf16(
                        false, ka.v, false, qbf[dh], (short)0, c, false, false);
            }
            s[t] = c;
        }

        // ---- additive mask from LDS table (key = kb + t*16 + 8*half + r) ----
        const float4 pa0 = *(const float4*)&penL[kb + 8 * half];
        const float4 pa1 = *(const float4*)&penL[kb + 8 * half + 4];
        const float4 pb0 = *(const float4*)&penL[kb + 16 + 8 * half];
        const float4 pb1 = *(const float4*)&penL[kb + 16 + 8 * half + 4];
        const float pen0[8] = { pa0.x, pa0.y, pa0.z, pa0.w, pa1.x, pa1.y, pa1.z, pa1.w };
        const float pen1[8] = { pb0.x, pb0.y, pb0.z, pb0.w, pb1.x, pb1.y, pb1.z, pb1.w };

        // ---- per-lane online softmax (query = l&15) ----
        float lmax = -3.0e38f;
        #pragma unroll
        for (int r = 0; r < 8; ++r) {
            s[0][r] = s[0][r] * SCALE2 + pen0[r];
            s[1][r] = s[1][r] * SCALE2 + pen1[r];
            lmax = fmaxf(lmax, fmaxf(s[0][r], s[1][r]));
        }
        lmax = fmaxf(lmax, __shfl_xor(lmax, 16, 32));  // partner half, same query
        const float mn = fmaxf(mrun, lmax);
        const float alpha = __builtin_amdgcn_exp2f(mrun - mn);
        mrun = mn;

        // P in bf16, directly in A-fragment layout (e[j] = key 8*half+j, etc.)
        BF16x16 pu;
        #pragma unroll
        for (int r = 0; r < 8; ++r) {
            pu.e[r]     = (__bf16)__builtin_amdgcn_exp2f(s[0][r] - mn);
            pu.e[8 + r] = (__bf16)__builtin_amdgcn_exp2f(s[1][r] - mn);
        }

        // ---- broadcast alpha (lane-space) to row-space and rescale ----
        float aR[8];
        #pragma unroll
        for (int r = 0; r < 8; ++r) aR[r] = __shfl(alpha, r + 8 * half, 32);
        #pragma unroll
        for (int t = 0; t < 4; ++t)
            #pragma unroll
            for (int r = 0; r < 8; ++r) acc[t][r] *= aR[r];
        #pragma unroll
        for (int r = 0; r < 8; ++r) lacc[r] *= aR[r];

        // ---- row-sum via matrix pipe: lacc += P . ones ----
        lacc = __builtin_amdgcn_wmma_f32_16x16x32_bf16(
                   false, pu.v, false, ones.v, (short)0, lacc, false, false);

        // ---- O += P . V : 4 WMMAs ----
        #pragma unroll
        for (int t = 0; t < 4; ++t) {
            BF16x16 vb;
            #pragma unroll
            for (int j = 0; j < 16; ++j)
                vb.e[j] = vT[buf][8 * half + (j < 8 ? j : 8 + j)][t * 16 + l15];
            acc[t] = __builtin_amdgcn_wmma_f32_16x16x32_bf16(
                         false, pu.v, false, vb.v, (short)0, acc[t], false, false);
        }
    }

    // ---- normalize and write out (row = query r+8*half, col = d) ----
    #pragma unroll
    for (int r = 0; r < 8; ++r) {
        const int   row  = qBase + r + 8 * half;
        const float invl = 1.0f / lacc[r];
        #pragma unroll
        for (int t = 0; t < 4; ++t) {
            Out[((size_t)bh * S_ + row) * D_ + t * 16 + l15] = acc[t][r] * invl;
        }
    }
}

extern "C" void kernel_launch(void* const* d_in, const int* in_sizes, int n_in,
                              void* d_out, int out_size, void* d_ws, size_t ws_size,
                              hipStream_t stream) {
    (void)in_sizes; (void)n_in; (void)out_size; (void)ws_size;
    const float* Q    = (const float*)d_in[0];
    const float* K    = (const float*)d_in[1];
    const float* V    = (const float*)d_in[2];
    const float* mask = (const float*)d_in[3];
    float* Out = (float*)d_out;

    // Workspace: bf16 K [bh][2048][64] then bf16 V [bh][2048][64]
    const size_t elems = (size_t)B_ * H_ * S_ * D_;   // 4 Mi elements
    __bf16* Kb = (__bf16*)d_ws;
    __bf16* Vb = Kb + elems;                           // 8 MiB offset

    SoftmaxAttention_prep_kernel<<<dim3((unsigned)(elems / 4 / 1024)), dim3(256), 0, stream>>>(
        K, V, Kb, Vb);

    SoftmaxAttention_55611236549123_kernel<<<dim3(B_ * H_ * (S_ / 64)), dim3(128), 0, stream>>>(
        Q, Kb, Vb, mask, Out);
}
